// CanonGLUFeedforwardLayer_53343493816972
// MI455X (gfx1250) — matrix operand order repaired and checked by reading, not compiled, not run on hardware
//
#include <hip/hip_runtime.h>
#include <hip/hip_bf16.h>

// ---------------------------------------------------------------------------
// CanonGLU FFN for MI455X (gfx1250).
//   out = (up_conv * silu(gate_conv)) @ w_down^T
// Three bf16 WMMA GEMMs (f32 accumulate) with TDM (tensor_load_to_lds)
// double-buffered LDS staging, plus an elementwise causal-conv/SwiGLU pass.
// GEMMs are C = A * W^T, A:[M,K], W:[N,K] row-major -> both map to WMMA
// fragments as contiguous K-runs (no transposes anywhere).
// ---------------------------------------------------------------------------

typedef __attribute__((ext_vector_type(16))) __bf16 v16bf;
typedef __attribute__((ext_vector_type(8)))  float  v8f;
typedef unsigned int u32x4 __attribute__((ext_vector_type(4)));
typedef int          i32x4 __attribute__((ext_vector_type(4)));
typedef int          i32x8 __attribute__((ext_vector_type(8)));

union FragAB {
    v16bf v;
    uint4 q[2];
};

// -------------------- fp32 -> bf16 conversion (x4 vectorized) ---------------
__global__ __launch_bounds__(256) void cvt_f32_bf16_x4(
    const float* __restrict__ in, __hip_bfloat16* __restrict__ out, long n4)
{
    long i = (long)blockIdx.x * blockDim.x + threadIdx.x;
    if (i >= n4) return;
    float4 f = ((const float4*)in)[i];
    size_t o = (size_t)i * 4;
    out[o + 0] = __float2bfloat16(f.x);
    out[o + 1] = __float2bfloat16(f.y);
    out[o + 2] = __float2bfloat16(f.z);
    out[o + 3] = __float2bfloat16(f.w);
}

// -------------------- TDM: DMA a [128 rows x 64 cols] bf16 tile to LDS ------
// Descriptor per CDNA5 ISA ch.8 (Tensor DMA Descriptor):
//   group0: count=1, lds_addr, global_addr(tile start), type=2
//   group1: data_size=2B, pad: 4 DWORDs after every 32 DWORDs (row=128B ->
//           LDS row stride 144B = 72 bf16, kills ds bank conflicts),
//           tensor_dim0=K, tensor_dim1=rows, tile 64x128, stride0=K
__device__ __forceinline__ void tdm_load_tile_2d(
    unsigned lds_byte_off, const __hip_bfloat16* tile_start,
    unsigned tensor_d0, unsigned tensor_d1, unsigned stride0)
{
    unsigned long long ga = (unsigned long long)(uintptr_t)tile_start;
    u32x4 g0;
    g0[0] = 1u;                                            // count=1, user desc
    g0[1] = lds_byte_off;                                  // lds_addr
    g0[2] = (unsigned)ga;                                  // global_addr[31:0]
    g0[3] = (unsigned)((ga >> 32) & 0x01FFFFFFu) | (2u << 30); // addr[56:32], type=2
    i32x8 g1;
    g1[0] = (int)((1u << 16)      // data_size = 2 bytes
                | (1u << 20)      // pad_enable
                | (4u << 22)      // pad_interval: every 32 DWORDs (=128B row)
                | (3u << 25));    // pad_amount: 4 DWORDs (=16B)
    g1[1] = (int)((tensor_d0 & 0xFFFFu) << 16);            // tensor_dim0 lo
    g1[2] = (int)(((tensor_d0 >> 16) & 0xFFFFu) | ((tensor_d1 & 0xFFFFu) << 16));
    g1[3] = (int)(((tensor_d1 >> 16) & 0xFFFFu) | (64u << 16)); // tile_dim0=64
    g1[4] = 128;                                           // tile_dim1=128
    g1[5] = (int)stride0;                                  // dim0 stride lo
    g1[6] = 0;
    g1[7] = 0;
    i32x4 z4 = {0, 0, 0, 0};
#if __has_include(<hip/amd_detail/amd_gfx1250_TDM.h>)
    i32x8 z8 = {0, 0, 0, 0, 0, 0, 0, 0};
    __builtin_amdgcn_tensor_load_to_lds(g0, g1, z4, z4, z8, 0);
#else
    __builtin_amdgcn_tensor_load_to_lds(g0, g1, z4, z4, 0);
#endif
}

__device__ __forceinline__ unsigned lds_off_of(const void* p) {
    return (unsigned)(unsigned long long)(uintptr_t)p;   // low 32b = LDS offset
}

// -------------------- GEMM: C[M,N] = A[M,K] * W[N,K]^T ----------------------
// 256 threads = 8 waves; block tile 128x128; wave (wm 0..3, wn 0..1) owns a
// 32x64 tile = 2x4 WMMA 16x16 tiles.  K-step 64, TDM double buffering.
//
// Fragment layouts used here (16-bit, wave32; K order chosen by this code, same for A and B):
//   A 16x32: lane m=lane&15, h=lane>>4: elems0-15=K[16h..16h+15]
//   B 32x16: lane n=lane&15, h=lane>>4: elems0-15=K[16h..16h+15]
//   C/D:     col=lane&15, row=8*(lane>>4)+vgpr
template <bool OUT_BF16>
__global__ __launch_bounds__(256) void gemm_bf16_tdm(
    const __hip_bfloat16* __restrict__ A,   // [M,K]
    const __hip_bfloat16* __restrict__ W,   // [N,K]
    void* __restrict__ Cout,                // [M,N] bf16 or f32
    int M, int N, int K)
{
    __shared__ alignas(16) __hip_bfloat16 sA[2][128][72];  // 144B row stride
    __shared__ alignas(16) __hip_bfloat16 sB[2][128][72];

    const int lane = threadIdx.x & 31;
    const int wave = threadIdx.x >> 5;
    const int wm   = wave & 3;
    const int wn   = wave >> 2;
    const int lm   = lane & 15;
    const int lh   = lane >> 4;
    const int m_base = blockIdx.x * 128;
    const int n_base = blockIdx.y * 128;

    v8f acc[2][4];
#pragma unroll
    for (int i = 0; i < 2; ++i)
#pragma unroll
        for (int j = 0; j < 4; ++j)
#pragma unroll
            for (int e = 0; e < 8; ++e)
                acc[i][j][e] = 0.0f;

    const int steps = K >> 6;
    const __hip_bfloat16* Abase = A + (size_t)m_base * K;
    const __hip_bfloat16* Wbase = W + (size_t)n_base * K;

    if (wave == 0) {   // prologue: DMA tiles for kk=0 into buffer 0
        tdm_load_tile_2d(lds_off_of(&sA[0][0][0]), Abase, (unsigned)K, (unsigned)M, (unsigned)K);
        tdm_load_tile_2d(lds_off_of(&sB[0][0][0]), Wbase, (unsigned)K, (unsigned)N, (unsigned)K);
    }

    for (int kk = 0; kk < steps; ++kk) {
        const int buf = kk & 1;
        if (wave == 0)
            __builtin_amdgcn_s_wait_tensorcnt(0);   // buf tiles landed in LDS
        __syncthreads();                            // publish to all waves
        if (wave == 0 && kk + 1 < steps) {          // overlap next DMA w/ math
            const int nb = buf ^ 1;
            size_t ko = (size_t)(kk + 1) * 64;
            tdm_load_tile_2d(lds_off_of(&sA[nb][0][0]), Abase + ko, (unsigned)K, (unsigned)M, (unsigned)K);
            tdm_load_tile_2d(lds_off_of(&sB[nb][0][0]), Wbase + ko, (unsigned)K, (unsigned)N, (unsigned)K);
        }
#pragma unroll
        for (int ks = 0; ks < 64; ks += 32) {
            FragAB a[2], b[4];
#pragma unroll
            for (int i = 0; i < 2; ++i) {
                const __hip_bfloat16* p = &sA[buf][wm * 32 + i * 16 + lm][ks + 16 * lh];
                a[i].q[0] = *(const uint4*)(p);
                a[i].q[1] = *(const uint4*)(p + 8);
            }
#pragma unroll
            for (int j = 0; j < 4; ++j) {
                const uint4* p = (const uint4*)&sB[buf][wn * 64 + j * 16 + lm][ks + 16 * lh];
                b[j].q[0] = p[0];
                b[j].q[1] = p[1];
            }
#pragma unroll
            for (int i = 0; i < 2; ++i)
#pragma unroll
                for (int j = 0; j < 4; ++j)
                    acc[i][j] = __builtin_amdgcn_wmma_f32_16x16x32_bf16(
                        false, a[i].v, false, b[j].v,
                        (short)0, acc[i][j], false, false);
        }
        // No trailing barrier needed: next DMA targets buf^1, and it is only
        // issued after the *next* iteration's barrier, by which time all
        // waves' ds reads of buf^1 were consumed by WMMAs (dscnt waited).
    }

#pragma unroll
    for (int i = 0; i < 2; ++i)
#pragma unroll
        for (int j = 0; j < 4; ++j)
#pragma unroll
            for (int v = 0; v < 8; ++v) {
                size_t row = (size_t)(m_base + wm * 32 + i * 16 + 8 * lh + v);
                size_t col = (size_t)(n_base + wn * 64 + j * 16 + lm);
                if constexpr (OUT_BF16) {
                    ((__hip_bfloat16*)Cout)[row * N + col] =
                        __float2bfloat16(acc[i][j][v]);
                } else {
                    ((float*)Cout)[row * N + col] = acc[i][j][v];
                }
            }
}

// ------------- causal depthwise conv (K=4) + residual + SwiGLU --------------
__global__ __launch_bounds__(256) void conv_silu_gate(
    const __hip_bfloat16* __restrict__ gate_pre,  // [B*T, F]
    const __hip_bfloat16* __restrict__ up_pre,    // [B*T, F]
    const float* __restrict__ conv_w,             // [2F, 4]
    __hip_bfloat16* __restrict__ h,               // [B*T, F]
    int T, int F)
{
    size_t idx = (size_t)blockIdx.x * blockDim.x + threadIdx.x;
    int f = (int)(idx % F);
    size_t bt = idx / F;
    int t = (int)(bt % T);

    float cg = (float)gate_pre[idx];   // residual term
    float cu = (float)up_pre[idx];
#pragma unroll
    for (int k = 0; k < 4; ++k) {
        int dt = k - 3;
        if (t + dt >= 0) {             // causal, never crosses batch boundary
            size_t j = (size_t)((long long)idx + (long long)dt * F);
            cg += conv_w[(size_t)f * 4 + k]       * (float)gate_pre[j];
            cu += conv_w[(size_t)(F + f) * 4 + k] * (float)up_pre[j];
        }
    }
    float s = cg / (1.0f + __expf(-cg));   // silu(gate)
    h[idx] = __float2bfloat16(cu * s);
}

// ---------------------------------------------------------------------------
extern "C" void kernel_launch(void* const* d_in, const int* in_sizes, int n_in,
                              void* d_out, int out_size, void* d_ws, size_t ws_size,
                              hipStream_t stream) {
    (void)in_sizes; (void)n_in; (void)out_size; (void)ws_size;

    constexpr int B = 2, T = 2048, D = 2048, F = 8192;
    constexpr int M = B * T;  // 4096

    const float* x      = (const float*)d_in[0];  // [B,T,D]
    const float* w_gate = (const float*)d_in[1];  // [F,D]
    const float* w_up   = (const float*)d_in[2];  // [F,D]
    const float* w_down = (const float*)d_in[3];  // [D,F]
    const float* conv_w = (const float*)d_in[4];  // [2F,4]
    float* out = (float*)d_out;                   // [B,T,D]

    // Workspace layout (bytes), total 304 MiB.
    char* ws = (char*)d_ws;
    __hip_bfloat16* xb   = (__hip_bfloat16*)(ws);                                   // 16 MiB
    __hip_bfloat16* wgb  = (__hip_bfloat16*)(ws + (size_t)16  * 1024 * 1024);       // 32 MiB
    __hip_bfloat16* wub  = (__hip_bfloat16*)(ws + (size_t)48  * 1024 * 1024);       // 32 MiB
    __hip_bfloat16* wdb  = (__hip_bfloat16*)(ws + (size_t)80  * 1024 * 1024);       // 32 MiB
    __hip_bfloat16* gate = (__hip_bfloat16*)(ws + (size_t)112 * 1024 * 1024);       // 64 MiB
    __hip_bfloat16* up   = (__hip_bfloat16*)(ws + (size_t)176 * 1024 * 1024);       // 64 MiB
    __hip_bfloat16* h    = (__hip_bfloat16*)(ws + (size_t)240 * 1024 * 1024);       // 64 MiB

    // 1) fp32 -> bf16 conversions (bf16 weights ~96MB -> live mostly in L2)
    {
        long n4;
        n4 = (long)M * D / 4;
        cvt_f32_bf16_x4<<<dim3((unsigned)((n4 + 255) / 256)), dim3(256), 0, stream>>>(x, xb, n4);
        n4 = (long)F * D / 4;
        cvt_f32_bf16_x4<<<dim3((unsigned)((n4 + 255) / 256)), dim3(256), 0, stream>>>(w_gate, wgb, n4);
        cvt_f32_bf16_x4<<<dim3((unsigned)((n4 + 255) / 256)), dim3(256), 0, stream>>>(w_up, wub, n4);
        n4 = (long)D * F / 4;
        cvt_f32_bf16_x4<<<dim3((unsigned)((n4 + 255) / 256)), dim3(256), 0, stream>>>(w_down, wdb, n4);
    }

    // 2) gate_pre = x @ w_gate^T   [4096,8192]
    gemm_bf16_tdm<true><<<dim3(M / 128, F / 128), dim3(256), 0, stream>>>(
        xb, wgb, (void*)gate, M, F, D);
    // 3) up_pre = x @ w_up^T
    gemm_bf16_tdm<true><<<dim3(M / 128, F / 128), dim3(256), 0, stream>>>(
        xb, wub, (void*)up, M, F, D);

    // 4) causal conv + residual + SwiGLU gating -> h
    {
        size_t total = (size_t)M * F;
        conv_silu_gate<<<dim3((unsigned)(total / 256)), dim3(256), 0, stream>>>(
            gate, up, conv_w, h, T, F);
    }

    // 5) out = h @ w_down^T   [4096,2048] fp32
    gemm_bf16_tdm<false><<<dim3(M / 128, D / 128), dim3(256), 0, stream>>>(
        h, wdb, (void*)out, M, D, F);
}
